// RelativeGlobalAttention_36155034698079
// MI455X (gfx1250) — matrix operand order repaired and checked by reading, not compiled, not run on hardware
//
#include <hip/hip_runtime.h>
#include <hip/hip_bf16.h>

// Problem constants (B=1)
#define SEQ    2048
#define DMODEL 1024
#define NH     16
#define DHEAD  64
#define LD_SC  2056   // scores row stride in floats (2048 + 8 pad, keeps 32B alignment)

typedef _Float16 v16h __attribute__((ext_vector_type(16)));
typedef _Float16 h8   __attribute__((ext_vector_type(8)));
typedef float    v8f  __attribute__((ext_vector_type(8)));

#define WMMA_F32_F16(a, b, c) \
  __builtin_amdgcn_wmma_f32_16x16x32_f16(false, (a), false, (b), (short)0, (c), false, false)

// Async global->LDS 16-byte copy (ASYNCcnt path). ldsoff = workgroup-relative LDS
// byte offset, gaddr = 64-bit global address. Per-lane like any VMEM op.
__device__ __forceinline__ void async_g2l_b128(unsigned ldsoff, unsigned long long gaddr) {
  asm volatile("global_load_async_to_lds_b128 %0, %1, off" :: "v"(ldsoff), "v"(gaddr) : "memory");
}
__device__ __forceinline__ void wait_asynccnt0() {
  asm volatile("s_wait_asynccnt 0x0" ::: "memory");
}

// A-matrix fragment (16x32 f16): lane L: M = L&15; lanes<16 hold K {0..7,16..23},
// lanes>=16 hold K {8..15,24..31}  (ISA 7.12.2, 16-bit A layout)
__device__ __forceinline__ v16h load_fragA(const _Float16* p, int ld, int row, int k0, int lane) {
  const int kb = (lane & 16) ? 8 : 0;
  const _Float16* q = p + (size_t)row * ld + k0 + kb;
  h8 lo = *(const h8*)q;
  h8 hi = *(const h8*)(q + 16);
  v16h a;
#pragma unroll
  for (int i = 0; i < 8; ++i) { a[i] = lo[i]; a[i + 8] = hi[i]; }
  return a;
}

// B-matrix fragment (32x16 f16): lane L: N = L&15; lanes<16 hold K {0..7,16..23},
// lanes>=16 hold K {8..15,24..31} (same K order as the A fragment).
// "row" = row index of the K-major source (i.e. column N of B), contiguous in K.
__device__ __forceinline__ v16h load_fragB(const _Float16* p, int ld, int row, int k0, int lane) {
  const int kb = (lane & 16) ? 8 : 0;
  const _Float16* q = p + (size_t)row * ld + k0 + kb;
  h8 lo = *(const h8*)q;
  h8 hi = *(const h8*)(q + 16);
  v16h a;
#pragma unroll
  for (int i = 0; i < 8; ++i) { a[i] = lo[i]; a[i + 8] = hi[i]; }
  return a;
}

// B fragment from an LDS-staged K-major tile (row stride 32 halves).
__device__ __forceinline__ v16h load_fragB_lds(const _Float16* tile, int row, int lane) {
  const int kb = (lane & 16) ? 8 : 0;
  const _Float16* q = tile + row * 32 + kb;
  v16h b;
#pragma unroll
  for (int i = 0; i < 16; ++i) b[i] = q[i < 8 ? i : i + 8];
  return b;
}

// ---------------------------------------------------------------- convert f32 -> f16
__global__ void __launch_bounds__(256)
rga_cvt_kernel(const float* __restrict__ src, _Float16* __restrict__ dst, int n) {
  int i = blockIdx.x * 256 + threadIdx.x;
  if (i < n) dst[i] = (_Float16)src[i];
}

// ---------------------------------------------------------------- QKV projection
// out = X @ W^T + bias ; stored head-major: transposeV=0 -> dst[h][s][dk]
//                                           transposeV=1 -> dst[h][dk][s]
// The 64x32 W tile is shared by all 4 waves of the block: stage it once per
// K-step via async global->LDS copies, double-buffered against the WMMAs.
// All 4 B fragments are loaded before the 4 WMMAs so dscnt waits stay partial.
__global__ void __launch_bounds__(128)
rga_proj_kernel(const _Float16* __restrict__ X, const _Float16* __restrict__ W,
                const float* __restrict__ bias, _Float16* __restrict__ dst, int transposeV) {
  __shared__ _Float16 wbuf[2][64 * 32];   // 2 x 4KB W tiles

  const int tid  = threadIdx.x;
  const int lane = tid & 31;
  const int w    = tid >> 5;
  const int c    = lane & 15;
  const int hi   = (lane >> 4) & 1;
  const int s0   = blockIdx.x * 64 + w * 16;
  const int n0   = blockIdx.y * 64;

  // stage W[n0 .. n0+63][k0 .. k0+31] into wbuf[buf]; 256 x 16B chunks / 128 threads
  auto stage = [&](int buf, int k0) {
#pragma unroll
    for (int rep = 0; rep < 2; ++rep) {
      const int idx = tid + rep * 128;        // 0..255
      const int row = idx >> 2, ch = idx & 3; // row 0..63, 16B chunk 0..3
      const unsigned long long g =
          (unsigned long long)(uintptr_t)(W + (size_t)(n0 + row) * DMODEL + k0 + ch * 8);
      const unsigned l = (unsigned)(uintptr_t)(&wbuf[buf][row * 32 + ch * 8]);
      async_g2l_b128(l, g);
    }
  };

  v8f a0 = {}, a1 = {}, a2 = {}, a3 = {};
  stage(0, 0);
  int cur = 0;
  for (int k0 = 0; k0 < DMODEL; k0 += 32, cur ^= 1) {
    wait_asynccnt0();
    __syncthreads();
    if (k0 + 32 < DMODEL) stage(cur ^ 1, k0 + 32);

    v16h a  = load_fragA(X, DMODEL, s0 + c, k0, lane);
    v16h b0 = load_fragB_lds(wbuf[cur], 0  + c, lane);
    v16h b1 = load_fragB_lds(wbuf[cur], 16 + c, lane);
    v16h b2 = load_fragB_lds(wbuf[cur], 32 + c, lane);
    v16h b3 = load_fragB_lds(wbuf[cur], 48 + c, lane);
    a0 = WMMA_F32_F16(a, b0, a0);
    a1 = WMMA_F32_F16(a, b1, a1);
    a2 = WMMA_F32_F16(a, b2, a2);
    a3 = WMMA_F32_F16(a, b3, a3);
  }
#pragma unroll
  for (int t = 0; t < 4; ++t) {
    v8f acc = (t == 0) ? a0 : (t == 1) ? a1 : (t == 2) ? a2 : a3;
    const int n  = n0 + t * 16 + c;
    const float bn = bias[n];
    const int hh = n >> 6, dk = n & 63;
#pragma unroll
    for (int g = 0; g < 8; ++g) {
      const int srow = s0 + g + 8 * hi;
      const float v = acc[g] + bn;
      if (transposeV) dst[((size_t)hh * DHEAD + dk) * SEQ + srow] = (_Float16)v;
      else            dst[((size_t)hh * SEQ + srow) * DHEAD + dk] = (_Float16)v;
    }
  }
}

// ---------------------------------------------------------------- attention core
// One (head, 16-row block) per 128-thread workgroup.
// bias(i,j): d=j-i; d<=0 -> q_i . Er[S-1+d]; d==1 -> 0; d>=2 -> q_{i+1} . Er[d-2]
__global__ void __launch_bounds__(128)
rga_attn_kernel(const _Float16* __restrict__ Q16, const _Float16* __restrict__ K16,
                const _Float16* __restrict__ VT16, const _Float16* __restrict__ Er16,
                _Float16* __restrict__ Attn16) {
  extern __shared__ char smem_raw[];
  float* sc     = reinterpret_cast<float*>(smem_raw); // [16][LD_SC] scores panel
  float* strips = sc + 16 * LD_SC;                    // [4 waves][2][16][32]
  float* rinv   = strips + 4 * 2 * 16 * 32;           // [16] reciprocal row sums

  const int tid  = threadIdx.x;
  const int lane = tid & 31;
  const int w    = tid >> 5;
  const int c    = lane & 15;
  const int hi   = (lane >> 4) & 1;
  const int h    = blockIdx.y;
  const int i0   = blockIdx.x * 16;

  const _Float16* Qh = Q16 + (size_t)h * SEQ * DHEAD;
  const _Float16* Kh = K16 + (size_t)h * SEQ * DHEAD;
  const _Float16* Vh = VT16 + (size_t)h * DHEAD * SEQ;

  // Q rows i0..i0+15 and shifted rows i0+1..i0+16 (clamped; clamped rows unused)
  v16h qa0 = load_fragA(Qh, DHEAD, i0 + c, 0, lane);
  v16h qa1 = load_fragA(Qh, DHEAD, i0 + c, 32, lane);
  int srow = i0 + c + 1; if (srow > SEQ - 1) srow = SEQ - 1;
  v16h qs0 = load_fragA(Qh, DHEAD, srow, 0, lane);
  v16h qs1 = load_fragA(Qh, DHEAD, srow, 32, lane);

  float* mystrip = strips + w * (2 * 16 * 32);

  // ---- pass 1: scores = (q.k + skewed bias) / sqrt(dk) into LDS panel
  for (int jt = w; jt < SEQ / 16; jt += 4) {
    const int j0 = jt * 16;
    const int Dd = j0 - i0;

    v8f acc = {};
    {
      v16h b0 = load_fragB(Kh, DHEAD, j0 + c, 0, lane);
      v16h b1 = load_fragB(Kh, DHEAD, j0 + c, 32, lane);
      acc = WMMA_F32_F16(qa0, b0, acc);
      acc = WMMA_F32_F16(qa1, b1, acc);
    }

    if (Dd <= 0) {  // below/at diagonal strip: strip1[r][u] = q_{i0+r} . Er[tb1+u]
      const int tb = SEQ - 1 + Dd - 15;
#pragma unroll
      for (int half = 0; half < 2; ++half) {
        int row = tb + half * 16 + c;
        row = row < 0 ? 0 : (row > SEQ - 1 ? SEQ - 1 : row);
        v16h b0 = load_fragB(Er16, DHEAD, row, 0, lane);
        v16h b1 = load_fragB(Er16, DHEAD, row, 32, lane);
        v8f st = {};
        st = WMMA_F32_F16(qa0, b0, st);
        st = WMMA_F32_F16(qa1, b1, st);
#pragma unroll
        for (int g = 0; g < 8; ++g)
          mystrip[(g + 8 * hi) * 32 + half * 16 + c] = st[g];
      }
    }
    if (Dd >= 0) {  // above/at diagonal strip: strip2[r][u] = q_{i0+r+1} . Er[tb2+u]
      const int tb = Dd - 17;
#pragma unroll
      for (int half = 0; half < 2; ++half) {
        int row = tb + half * 16 + c;
        row = row < 0 ? 0 : (row > SEQ - 1 ? SEQ - 1 : row);
        v16h b0 = load_fragB(Er16, DHEAD, row, 0, lane);
        v16h b1 = load_fragB(Er16, DHEAD, row, 32, lane);
        v8f st = {};
        st = WMMA_F32_F16(qs0, b0, st);
        st = WMMA_F32_F16(qs1, b1, st);
#pragma unroll
        for (int g = 0; g < 8; ++g)
          mystrip[512 + (g + 8 * hi) * 32 + half * 16 + c] = st[g];
      }
    }

#pragma unroll
    for (int g = 0; g < 8; ++g) {
      const int r = g + 8 * hi;
      const int d = Dd + c - r;
      const int u = c - r + 15;           // u in [0,30]
      float bias = 0.0f;
      if (d != 1) bias = mystrip[(d <= 0 ? 0 : 512) + r * 32 + u];
      sc[r * LD_SC + j0 + c] = (acc[g] + bias) * 0.125f;
    }
  }
  __syncthreads();

  // ---- pass 2: row softmax (store un-normalized exp; keep 1/sum)
  for (int r = w; r < 16; r += 4) {
    float* row = sc + r * LD_SC;
    float mx = -3.4e38f;
    for (int t = lane; t < SEQ; t += 32) mx = fmaxf(mx, row[t]);
#pragma unroll
    for (int o = 16; o > 0; o >>= 1) mx = fmaxf(mx, __shfl_xor(mx, o, 32));
    float sum = 0.0f;
    for (int t = lane; t < SEQ; t += 32) {
      float e = __expf(row[t] - mx);
      row[t] = e;
      sum += e;
    }
#pragma unroll
    for (int o = 16; o > 0; o >>= 1) sum += __shfl_xor(sum, o, 32);
    if (lane == 0) rinv[r] = 1.0f / sum;
  }
  __syncthreads();

  // ---- pass 3: O = P @ V  (V^T layout makes B-fragments K-contiguous)
  v8f o0 = {}, o1 = {}, o2 = {}, o3 = {};
  for (int kt = w; kt < SEQ / 32; kt += 4) {
    const int j0 = kt * 32;
    v16h pa;
    {
      const int kb = hi ? 8 : 0;
      const float* p0 = sc + c * LD_SC + j0 + kb;
#pragma unroll
      for (int i = 0; i < 8; ++i) {
        pa[i]     = (_Float16)p0[i];
        pa[i + 8] = (_Float16)p0[16 + i];
      }
    }
    v16h b0 = load_fragB(Vh, SEQ, 0  + c, j0, lane);
    v16h b1 = load_fragB(Vh, SEQ, 16 + c, j0, lane);
    v16h b2 = load_fragB(Vh, SEQ, 32 + c, j0, lane);
    v16h b3 = load_fragB(Vh, SEQ, 48 + c, j0, lane);
    o0 = WMMA_F32_F16(pa, b0, o0);
    o1 = WMMA_F32_F16(pa, b1, o1);
    o2 = WMMA_F32_F16(pa, b2, o2);
    o3 = WMMA_F32_F16(pa, b3, o3);
  }
  __syncthreads();              // everyone done reading the scores panel

  float* op = sc;               // reuse panel: [4 waves][16][64] partial O
#pragma unroll
  for (int g = 0; g < 8; ++g) {
    const int r = g + 8 * hi;
    float* dst = op + (w * 16 + r) * 64 + c;
    dst[0]  = o0[g];
    dst[16] = o1[g];
    dst[32] = o2[g];
    dst[48] = o3[g];
  }
  __syncthreads();

  for (int e = tid; e < 16 * 64; e += 128) {
    const int r = e >> 6, cd = e & 63;
    float v = op[r * 64 + cd] + op[(16 + r) * 64 + cd] +
              op[(32 + r) * 64 + cd] + op[(48 + r) * 64 + cd];
    v *= rinv[r];
    Attn16[(size_t)(i0 + r) * DMODEL + h * DHEAD + cd] = (_Float16)v;
  }
}

// ---------------------------------------------------------------- output projection
__global__ void __launch_bounds__(128)
rga_outproj_kernel(const _Float16* __restrict__ X, const _Float16* __restrict__ W,
                   const float* __restrict__ bias, float* __restrict__ out) {
  __shared__ _Float16 wbuf[2][64 * 32];

  const int tid  = threadIdx.x;
  const int lane = tid & 31;
  const int w    = tid >> 5;
  const int c    = lane & 15;
  const int hi   = (lane >> 4) & 1;
  const int s0   = blockIdx.x * 64 + w * 16;
  const int n0   = blockIdx.y * 64;

  auto stage = [&](int buf, int k0) {
#pragma unroll
    for (int rep = 0; rep < 2; ++rep) {
      const int idx = tid + rep * 128;
      const int row = idx >> 2, ch = idx & 3;
      const unsigned long long g =
          (unsigned long long)(uintptr_t)(W + (size_t)(n0 + row) * DMODEL + k0 + ch * 8);
      const unsigned l = (unsigned)(uintptr_t)(&wbuf[buf][row * 32 + ch * 8]);
      async_g2l_b128(l, g);
    }
  };

  v8f a0 = {}, a1 = {}, a2 = {}, a3 = {};
  stage(0, 0);
  int cur = 0;
  for (int k0 = 0; k0 < DMODEL; k0 += 32, cur ^= 1) {
    wait_asynccnt0();
    __syncthreads();
    if (k0 + 32 < DMODEL) stage(cur ^ 1, k0 + 32);

    v16h a  = load_fragA(X, DMODEL, s0 + c, k0, lane);
    v16h b0 = load_fragB_lds(wbuf[cur], 0  + c, lane);
    v16h b1 = load_fragB_lds(wbuf[cur], 16 + c, lane);
    v16h b2 = load_fragB_lds(wbuf[cur], 32 + c, lane);
    v16h b3 = load_fragB_lds(wbuf[cur], 48 + c, lane);
    a0 = WMMA_F32_F16(a, b0, a0);
    a1 = WMMA_F32_F16(a, b1, a1);
    a2 = WMMA_F32_F16(a, b2, a2);
    a3 = WMMA_F32_F16(a, b3, a3);
  }
#pragma unroll
  for (int t = 0; t < 4; ++t) {
    v8f acc = (t == 0) ? a0 : (t == 1) ? a1 : (t == 2) ? a2 : a3;
    const int n = n0 + t * 16 + c;
    const float bn = bias[n];
#pragma unroll
    for (int g = 0; g < 8; ++g) {
      const int srow = s0 + g + 8 * hi;
      out[(size_t)srow * DMODEL + n] = acc[g] + bn;
    }
  }
}

// ---------------------------------------------------------------- host launcher
extern "C" void kernel_launch(void* const* d_in, const int* in_sizes, int n_in,
                              void* d_out, int out_size, void* d_ws, size_t ws_size,
                              hipStream_t stream) {
  (void)in_sizes; (void)n_in; (void)out_size; (void)ws_size;
  const float* query = (const float*)d_in[0];
  const float* key   = (const float*)d_in[1];
  const float* value = (const float*)d_in[2];
  const float* Wq    = (const float*)d_in[3];
  const float* bq    = (const float*)d_in[4];
  const float* Wk    = (const float*)d_in[5];
  const float* bk    = (const float*)d_in[6];
  const float* Wv    = (const float*)d_in[7];
  const float* bv    = (const float*)d_in[8];
  const float* Wo    = (const float*)d_in[9];
  const float* bo    = (const float*)d_in[10];
  const float* Er    = (const float*)d_in[11];

  _Float16* ws = (_Float16*)d_ws;
  size_t off = 0;
  auto alloc = [&](size_t elems) { _Float16* p = ws + off; off += elems; return p; };
  _Float16* Xq16 = alloc((size_t)SEQ * DMODEL);
  _Float16* Xk16 = alloc((size_t)SEQ * DMODEL);
  _Float16* Xv16 = alloc((size_t)SEQ * DMODEL);
  _Float16* Wq16 = alloc((size_t)DMODEL * DMODEL);
  _Float16* Wk16 = alloc((size_t)DMODEL * DMODEL);
  _Float16* Wv16 = alloc((size_t)DMODEL * DMODEL);
  _Float16* Wo16 = alloc((size_t)DMODEL * DMODEL);
  _Float16* Er16 = alloc((size_t)SEQ * DHEAD);
  _Float16* Q16  = alloc((size_t)NH * SEQ * DHEAD);
  _Float16* K16  = alloc((size_t)NH * SEQ * DHEAD);
  _Float16* VT16 = alloc((size_t)NH * SEQ * DHEAD);
  _Float16* At16 = alloc((size_t)SEQ * DMODEL);

  auto cvt = [&](const float* s, _Float16* d, int n) {
    rga_cvt_kernel<<<dim3((n + 255) / 256), dim3(256), 0, stream>>>(s, d, n);
  };
  cvt(query, Xq16, SEQ * DMODEL);
  cvt(key,   Xk16, SEQ * DMODEL);
  cvt(value, Xv16, SEQ * DMODEL);
  cvt(Wq, Wq16, DMODEL * DMODEL);
  cvt(Wk, Wk16, DMODEL * DMODEL);
  cvt(Wv, Wv16, DMODEL * DMODEL);
  cvt(Wo, Wo16, DMODEL * DMODEL);
  cvt(Er, Er16, SEQ * DHEAD);

  dim3 pg(SEQ / 64, DMODEL / 64);
  rga_proj_kernel<<<pg, dim3(128), 0, stream>>>(Xq16, Wq16, bq, Q16, 0);
  rga_proj_kernel<<<pg, dim3(128), 0, stream>>>(Xk16, Wk16, bk, K16, 0);
  rga_proj_kernel<<<pg, dim3(128), 0, stream>>>(Xv16, Wv16, bv, VT16, 1);

  const int smemBytes = (16 * LD_SC + 4 * 2 * 16 * 32 + 16) * (int)sizeof(float);
  hipFuncSetAttribute(reinterpret_cast<const void*>(rga_attn_kernel),
                      hipFuncAttributeMaxDynamicSharedMemorySize, smemBytes);
  rga_attn_kernel<<<dim3(SEQ / 16, NH), dim3(128), smemBytes, stream>>>(
      Q16, K16, VT16, Er16, At16);

  rga_outproj_kernel<<<pg, dim3(128), 0, stream>>>(At16, Wo16, bo, (float*)d_out);
}
